// WaveNet_32787780337997
// MI455X (gfx1250) — matrix operand order repaired and checked by reading, not compiled, not run on hardware
//
#include <hip/hip_runtime.h>
#include <hip/hip_bf16.h>
#include <cmath>

// ---- problem constants (match reference) ----
#define B_SZ      8
#define C_CH      128
#define T_LEN     8192
#define N_LAYERS  16
#define LC        (N_LAYERS * C_CH)   // 2048 skip rows
#define TILE_T    64
#define HALO_MAX  128
#define CP        136                  // padded LDS row stride in bf16 elems (272B = 68 DWORDs)
#define NFRAG     12                   // 8 hidden (2 taps x 4 ktiles) + 4 res ktiles

#ifndef HAVE_TDM
#if defined(__has_builtin)
#if __has_builtin(__builtin_amdgcn_tensor_load_to_lds)
#define HAVE_TDM 1
#endif
#endif
#endif
#ifndef HAVE_TDM
#define HAVE_TDM 0
#endif

typedef __attribute__((ext_vector_type(16))) __bf16 v16bf;
typedef __attribute__((ext_vector_type(8)))  float  v8f;
typedef __attribute__((ext_vector_type(2)))  float  f32x2;
typedef __attribute__((ext_vector_type(2)))  __bf16 bf16x2;
typedef __attribute__((ext_vector_type(4)))  unsigned int tdm_u4;
typedef __attribute__((ext_vector_type(8)))  int          tdm_i8;
typedef __attribute__((ext_vector_type(4)))  int          tdm_i4;

union BFrag {
  v16bf v;
  unsigned short h[16];
  uint4 q[2];
};
union CFrag {
  v8f   v;
  float f[8];
};

// native hardware bf16 conversions (RNE)
__device__ __forceinline__ unsigned short f2bf(float f) {
  union { __bf16 h; unsigned short u; } r;
  r.h = (__bf16)f;
  return r.u;
}
// pairwise pack -> single v_cvt_pk_bf16_f32
__device__ __forceinline__ unsigned pk2(float a, float b) {
  const f32x2 s = {a, b};
  union { bf16x2 h; unsigned u; } r;
  r.h = __builtin_convertvector(s, bf16x2);
  return r.u;
}

// ---------------------------------------------------------------------------
// Pre-pass 1: x [B,C,T] f32 -> xT [B,T,C] f32 (residual stream) and
//             xBF [B,T,C] bf16 (GEMM input stream, TDM-stageable)
// ---------------------------------------------------------------------------
__global__ __launch_bounds__(256)
void transpose_x_kernel(const float* __restrict__ x, float* __restrict__ xT,
                        unsigned short* __restrict__ xBF) {
  __shared__ float tile[32][33];
  const int b  = blockIdx.z;
  const int t0 = blockIdx.x * 32;
  const int c0 = blockIdx.y * 32;
  const int tx = threadIdx.x;          // 0..31
  const int ty = threadIdx.y;          // 0..7
  #pragma unroll
  for (int i = ty; i < 32; i += 8)
    tile[i][tx] = x[((size_t)b * C_CH + c0 + i) * T_LEN + t0 + tx];
  __syncthreads();
  #pragma unroll
  for (int i = ty; i < 32; i += 8) {
    const float v = tile[tx][i];
    const size_t off = ((size_t)b * T_LEN + t0 + i) * C_CH + c0 + tx;
    xT[off]  = v;
    xBF[off] = f2bf(v);
  }
}

// ---------------------------------------------------------------------------
// Pre-pass 2: shuffle all layers' weights into bf16 WMMA A-operand fragments.
//   fid 0..7  : hidden tap j=fid/4, ktile kt=fid%4;  fid 8..11 : res ktile
// A 16x32 bf16, contiguous k order used here (B matches): lane holds row M=lane&15; elem e -> K=e+16*(lane>>4)
// ---------------------------------------------------------------------------
__global__ __launch_bounds__(256)
void shuffle_w_kernel(const float* __restrict__ hw,   // [L,C,C,2]
                      const float* __restrict__ rw,   // [L,C,C]
                      unsigned short* __restrict__ wsW) {
  const int gid  = blockIdx.x * 256 + threadIdx.x;    // ((l*12+fid)*8+m)*32+lane
  const int lane = gid & 31;
  const int m    = (gid >> 5) & 7;
  const int lf   = gid >> 8;
  const int fid  = lf % NFRAG;
  const int l    = lf / NFRAG;
  if (l >= N_LAYERS) return;
  const int half  = lane >> 4;
  const int c_out = m * 16 + (lane & 15);
  unsigned o2[8];
  #pragma unroll
  for (int e2 = 0; e2 < 8; ++e2) {
    float v[2];
    #pragma unroll
    for (int k = 0; k < 2; ++k) {
      const int e = e2 * 2 + k;
      const int cin = ((fid & 3) * 32) + e + half * 16;
      if (fid < 8) {
        const int j = fid >> 2;
        v[k] = hw[(((size_t)l * C_CH + c_out) * C_CH + cin) * 2 + j];
      } else {
        v[k] = rw[((size_t)l * C_CH + c_out) * C_CH + cin];
      }
    }
    o2[e2] = pk2(v[0], v[1]);
  }
  uint4* dst = (uint4*)&wsW[(size_t)gid * 16];
  uint4 q0, q1;
  q0.x = o2[0]; q0.y = o2[1]; q0.z = o2[2]; q0.w = o2[3];
  q1.x = o2[4]; q1.y = o2[5]; q1.z = o2[6]; q1.w = o2[7];
  dst[0] = q0;
  dst[1] = q1;
}

// ---------------------------------------------------------------------------
// One WaveNet layer. f32 residual stream + bf16 GEMM stream, both [B,T,C].
// ---------------------------------------------------------------------------
__global__ __launch_bounds__(256)
void wavenet_layer_kernel(const float*          __restrict__ xinT,   // [B,T,C] f32
                          const unsigned short* __restrict__ xinBF,  // [B,T,C] bf16
                          float*                __restrict__ xoutT,   // [B,T,C] f32
                          unsigned short*       __restrict__ xoutBF,  // [B,T,C] bf16
                          const unsigned short* __restrict__ wfrag,   // layer frags
                          const float* __restrict__ hb,     // [C]
                          const float* __restrict__ rb,     // [C]
                          float*       __restrict__ skip,   // d_out + l*C*T
                          int dil) {
  __shared__ unsigned short lds_x[(TILE_T + HALO_MAX) * CP]; // bf16 [t][c], row stride 68 DW
  __shared__ unsigned short lds_g[2][16 * CP];               // bf16 gated, double-buffered

  const int tid  = threadIdx.x;
  const int b    = blockIdx.y;
  const int t0   = blockIdx.x * TILE_T;
  const int m    = tid >> 5;     // wave -> C_out tile
  const int lane = tid & 31;
  const int half = lane >> 4;
  const int ln   = lane & 15;

  const int span   = TILE_T + dil;           // rows to stage, <= 192
  const int tstart = t0 - dil;

  // ---- stage xBF[tstart .. t0+TILE_T) into LDS as bf16 [t][c] ----
#if HAVE_TDM
  if (tstart >= 0) {
    // Tensor Data Mover: one DMA for the whole padded tile. Wave 0 issues it.
    if (tid < 32) {
      const unsigned lds_base =
          (unsigned)(unsigned long long)(__attribute__((address_space(3))) unsigned short*)lds_x;
      const unsigned long long ga =
          (unsigned long long)(const void*)(xinBF + ((size_t)b * T_LEN + tstart) * C_CH);
      tdm_u4 g0;
      g0[0] = 1u;                                    // count=1, user descriptor
      g0[1] = lds_base;                              // LDS byte address
      g0[2] = (unsigned)(ga & 0xFFFFFFFFu);          // global_addr[31:0]
      g0[3] = (unsigned)((ga >> 32) & 0x01FFFFFFu) | (2u << 30);  // addr[56:32] | type=2
      const int dim1 = T_LEN - tstart;               // rows available (OOB bound)
      tdm_i8 g1;
      g1[0] = (int)((1u << 16)      // data_size = 2 bytes
                  | (1u << 20)      // pad_enable
                  | (5u << 22)      // pad_interval: every 64 DWORDs (=128 bf16 row)
                  | (3u << 25));    // pad_amount: 4 DWORDs (row stride -> 68 DW = CP)
      g1[1] = (int)(128u << 16);                     // tensor_dim0 = 128 elems
      g1[2] = (int)(((unsigned)(dim1 & 0xFFFF)) << 16);   // tensor_dim1 lo
      g1[3] = (int)(((unsigned)dim1 >> 16) & 0xFFFFu) | (128 << 16); // dim1 hi | tile_dim0=128
      g1[4] = span & 0xFFFF;                         // tile_dim1 = rows, tile_dim2 = 0
      g1[5] = 128;                                   // tensor_dim0_stride = 128 elems
      g1[6] = 0;
      g1[7] = 0;
      const tdm_i4 gz = {0, 0, 0, 0};
#if __clang_major__ >= 23
      const tdm_i8 gz8 = {0, 0, 0, 0, 0, 0, 0, 0};
      __builtin_amdgcn_tensor_load_to_lds(g0, g1, gz, gz, gz8, 0);
#else
      __builtin_amdgcn_tensor_load_to_lds(g0, g1, gz, gz, 0);
#endif
      __builtin_amdgcn_s_wait_tensorcnt(0);
    }
  } else
#endif
  {
    // manual path: boundary tiles (zero-fill t<0) or no-TDM toolchain
    const int rlane = tid & 15;              // 8-channel group within a row
    const int rrow  = tid >> 4;              // 16 rows per pass
    for (int tr = rrow; tr < span; tr += 16) {
      const int tg = tstart + tr;
      uint4 q = {0u, 0u, 0u, 0u};
      if (tg >= 0)
        q = *(const uint4*)&xinBF[((size_t)b * T_LEN + tg) * C_CH + (rlane << 3)];
      *(uint4*)&lds_x[tr * CP + (rlane << 3)] = q;
    }
  }

  // ---- load this wave's pre-shuffled A fragments (coalesced b128 pairs) ----
  BFrag Ah[2][4];
  BFrag Ar[4];
  {
    const uint4* wf = (const uint4*)wfrag;   // frag f: idx ((f*8+m)*32+lane)*2
    #pragma unroll
    for (int f = 0; f < 8; ++f) {
      const size_t i = ((size_t)(f * 8 + m) * 32 + lane) * 2;
      Ah[f >> 2][f & 3].q[0] = wf[i];
      Ah[f >> 2][f & 3].q[1] = wf[i + 1];
    }
    #pragma unroll
    for (int kt = 0; kt < 4; ++kt) {
      const size_t i = ((size_t)((8 + kt) * 8 + m) * 32 + lane) * 2;
      Ar[kt].q[0] = wf[i];
      Ar[kt].q[1] = wf[i + 1];
    }
  }

  // ---- biases: 8 consecutive channels per lane (contiguous 32B) ----
  const int c0 = m * 16 + half * 8;          // C/D layout: elem r -> channel c0 + r
  float4 hb0 = *(const float4*)&hb[c0];
  float4 hb1 = *(const float4*)&hb[c0 + 4];
  float4 rb0 = *(const float4*)&rb[c0];
  float4 rb1 = *(const float4*)&rb[c0 + 4];

  __syncthreads();

  // ---- main loop over 16-wide time tiles ----
  #pragma unroll 1
  for (int nt = 0; nt < TILE_T / 16; ++nt) {
    const int tg = t0 + nt * 16 + ln;        // this lane's time column

    // hidden conv: h = W0 @ x(t-d) + W1 @ x(t) + b
    CFrag acc;
    acc.f[0] = hb0.x; acc.f[1] = hb0.y; acc.f[2] = hb0.z; acc.f[3] = hb0.w;
    acc.f[4] = hb1.x; acc.f[5] = hb1.y; acc.f[6] = hb1.z; acc.f[7] = hb1.w;

    #pragma unroll
    for (int j = 0; j < 2; ++j) {            // tap 0 -> x(t-d), tap 1 -> x(t)
      const int row = nt * 16 + ln + j * dil;
      const uint4* src = (const uint4*)&lds_x[row * CP];
      #pragma unroll
      for (int kt = 0; kt < 4; ++kt) {
        // B 32x16 bf16, this code's contiguous k order (matches A): lane = col N; elems = K in [kt*32+16*half, +16)
        BFrag Bx;
        Bx.q[0] = src[kt * 4 + half * 2];
        Bx.q[1] = src[kt * 4 + half * 2 + 1];
        acc.v = __builtin_amdgcn_wmma_f32_16x16x32_bf16(
            false, Ah[j][kt].v, false, Bx.v, (short)0, acc.v, false, false);
      }
    }

    // gate; stream skip to HBM (non-temporal); stage gated bf16 in LDS
    float gv[8];
    #pragma unroll
    for (int r = 0; r < 8; ++r) {
      const float h = acc.f[r];
      const float g = tanhf(h) * (1.0f / (1.0f + __expf(-h)));
      gv[r] = g;
      __builtin_nontemporal_store(g, &skip[((size_t)b * LC + c0 + r) * T_LEN + tg]);
    }
    {
      uint4 q;
      q.x = pk2(gv[0], gv[1]);  q.y = pk2(gv[2], gv[3]);
      q.z = pk2(gv[4], gv[5]);  q.w = pk2(gv[6], gv[7]);
      *(uint4*)&lds_g[nt & 1][ln * CP + c0] = q;
    }
    __syncthreads();

    // res 1x1 conv: r = Rw @ gated + rb
    CFrag racc;
    racc.f[0] = rb0.x; racc.f[1] = rb0.y; racc.f[2] = rb0.z; racc.f[3] = rb0.w;
    racc.f[4] = rb1.x; racc.f[5] = rb1.y; racc.f[6] = rb1.z; racc.f[7] = rb1.w;
    const uint4* gsrc = (const uint4*)&lds_g[nt & 1][ln * CP];
    #pragma unroll
    for (int kt = 0; kt < 4; ++kt) {
      BFrag Bg;
      Bg.q[0] = gsrc[kt * 4 + half * 2];
      Bg.q[1] = gsrc[kt * 4 + half * 2 + 1];
      racc.v = __builtin_amdgcn_wmma_f32_16x16x32_bf16(
          false, Ar[kt].v, false, Bg.v, (short)0, racc.v, false, false);
    }

    // residual add; write f32 stream + packed bf16 stream (L2-hot, 32B/16B per lane)
    {
      const size_t off = ((size_t)b * T_LEN + tg) * C_CH + c0;
      const float4 a0 = *(const float4*)&xinT[off];
      const float4 a1 = *(const float4*)&xinT[off + 4];
      float4 o0, o1;
      o0.x = racc.f[0] + a0.x;  o0.y = racc.f[1] + a0.y;
      o0.z = racc.f[2] + a0.z;  o0.w = racc.f[3] + a0.w;
      o1.x = racc.f[4] + a1.x;  o1.y = racc.f[5] + a1.y;
      o1.z = racc.f[6] + a1.z;  o1.w = racc.f[7] + a1.w;
      *(float4*)&xoutT[off] = o0;
      *(float4*)&xoutT[off + 4] = o1;
      uint4 qb;
      qb.x = pk2(o0.x, o0.y);  qb.y = pk2(o0.z, o0.w);
      qb.z = pk2(o1.x, o1.y);  qb.w = pk2(o1.z, o1.w);
      *(uint4*)&xoutBF[off] = qb;
    }
  }
}

// ---------------------------------------------------------------------------
extern "C" void kernel_launch(void* const* d_in, const int* in_sizes, int n_in,
                              void* d_out, int out_size, void* d_ws, size_t ws_size,
                              hipStream_t stream) {
  const float* x  = (const float*)d_in[0];   // [B,C,T]
  const float* hw = (const float*)d_in[1];   // [L,C,C,2]
  const float* hb = (const float*)d_in[2];   // [L,C]
  const float* rw = (const float*)d_in[3];   // [L,C,C,1]
  const float* rb = (const float*)d_in[4];   // [L,C]
  float* out = (float*)d_out;                // [B, L*C, T]

  const size_t act_elems = (size_t)B_SZ * C_CH * T_LEN;
  float* actA = (float*)d_ws;                          // f32 residual ping
  float* actB = actA + act_elems;                      // f32 residual pong
  unsigned short* bfA = (unsigned short*)(actB + act_elems);   // bf16 stream ping
  unsigned short* bfB = bfA + act_elems;                       // bf16 stream pong
  unsigned short* wsW = bfB + act_elems;                       // shuffled weight frags

  static const int dil[N_LAYERS] = {1, 2, 4, 8, 16, 32, 64, 128,
                                    1, 2, 4, 8, 16, 32, 64, 128};

  // pre-passes (tiny vs 536MB output stream)
  transpose_x_kernel<<<dim3(T_LEN / 32, C_CH / 32, B_SZ), dim3(32, 8), 0, stream>>>(
      x, actA, bfA);
  {
    const int total = N_LAYERS * NFRAG * 8 * 32;       // 49152 fragments
    shuffle_w_kernel<<<(total + 255) / 256, 256, 0, stream>>>(hw, rw, wsW);
  }

  dim3 grid(T_LEN / TILE_T, B_SZ);
  dim3 block(256);
  const float* curF = actA;
  const unsigned short* curB = bfA;
  for (int l = 0; l < N_LAYERS; ++l) {
    float*          nxtF = (l & 1) ? actA : actB;      // l=0: A -> B
    unsigned short* nxtB = (l & 1) ? bfA  : bfB;
    wavenet_layer_kernel<<<grid, block, 0, stream>>>(
        curF, curB, nxtF, nxtB,
        wsW + (size_t)l * NFRAG * 8 * 32 * 16,
        hb + (size_t)l * C_CH,
        rb + (size_t)l * C_CH,
        out + (size_t)l * C_CH * T_LEN,
        dil[l]);
    curF = nxtF;
    curB = nxtB;
  }
}
